// CrossAttensionFusion_51067161150175
// MI455X (gfx1250) — compile-verified, matrix operand order checked by reading, not run on hardware
//
#include <hip/hip_runtime.h>
#include <hip/hip_bf16.h>

typedef __bf16 bf16_t;
typedef __attribute__((ext_vector_type(8)))  bf16_t v8bf;
typedef __attribute__((ext_vector_type(16))) bf16_t v16bf;
typedef __attribute__((ext_vector_type(8)))  float  v8f;

#define E_DIM  384
#define B_DIM  512
#define H_DIM  512
#define TWO_E  768
#define NGROUP 32
#define GSIZE  12

// ---------------------------------------------------------------------------
// Convert + transpose weights to bf16: Wt[n*K + k] = bf16(W[k*N + n])
// ---------------------------------------------------------------------------
__global__ void convert_transpose_kernel(const float* __restrict__ W,
                                         bf16_t* __restrict__ Wt,
                                         int K, int N) {
    int idx = blockIdx.x * blockDim.x + threadIdx.x;
    if (idx >= K * N) return;
    int k = idx / N;
    int n = idx - k * N;
    Wt[(size_t)n * K + k] = (bf16_t)W[(size_t)k * N + n];
}

// ---------------------------------------------------------------------------
// GroupNorm: one wave per row, lane == group (32 groups of 12 channels).
// Writes bf16 activations for the WMMA GEMMs.
// ---------------------------------------------------------------------------
__global__ void groupnorm_kernel(const float* __restrict__ x,
                                 const float* __restrict__ xb,
                                 const float* __restrict__ gamma,
                                 const float* __restrict__ beta,
                                 const float* __restrict__ gamma_b,
                                 const float* __restrict__ beta_b,
                                 bf16_t* __restrict__ h,
                                 bf16_t* __restrict__ hb) {
    int row = blockIdx.x * blockDim.y + threadIdx.y;
    const float* xs; const float* gs; const float* bs; bf16_t* hs;
    if (blockIdx.y == 0) { xs = x;  gs = gamma;   bs = beta;   hs = h;  }
    else                 { xs = xb; gs = gamma_b; bs = beta_b; hs = hb; }

    int g = threadIdx.x;                       // group index 0..31
    const float* p = xs + (size_t)row * E_DIM + g * GSIZE;

    float s = 0.f;
#pragma unroll
    for (int t = 0; t < GSIZE; ++t) s += p[t];
    float mean = s * (1.0f / GSIZE);

    float var = 0.f;
#pragma unroll
    for (int t = 0; t < GSIZE; ++t) { float d = p[t] - mean; var += d * d; }
    var *= (1.0f / GSIZE);
    float inv = rsqrtf(var + 1e-6f);

#pragma unroll
    for (int t = 0; t < GSIZE; ++t) {
        int c = g * GSIZE + t;
        float y = (p[t] - mean) * inv * gs[c] + bs[c];
        hs[(size_t)row * E_DIM + c] = (bf16_t)y;
    }
}

// ---------------------------------------------------------------------------
// Fragment load matching the CDNA5 16-bit A/B VGPR layout: per lane, the 16
// K-values are {half*8 .. half*8+7} and {16+half*8 .. 16+half*8+7}.
// ---------------------------------------------------------------------------
__device__ __forceinline__ v16bf load_frag(const bf16_t* __restrict__ row,
                                           int k0, int half) {
    v8bf lo = *(const v8bf*)(row + k0 + half * 8);
    v8bf hi = *(const v8bf*)(row + k0 + 16 + half * 8);
    return __builtin_shufflevector(lo, hi,
               0,1,2,3,4,5,6,7,8,9,10,11,12,13,14,15);
}

// ---------------------------------------------------------------------------
// bf16 WMMA GEMM: C[M][N] = A[M][K] * Bt[N][K]^T + bias[N]  (f32 accumulate)
// Register-blocked: one wave computes a 32x32 tile (2 A-frags x 2 B-frags ->
// 4 WMMAs per k-step, 2 b128 loads per WMMA). Block = (32,4): 4 waves tile
// 128 columns; grid = (N/128, M/32).
// ---------------------------------------------------------------------------
__global__ void wmma_gemm_bf16(const bf16_t* __restrict__ A,
                               const bf16_t* __restrict__ Bt,
                               const float*  __restrict__ bias,
                               float* __restrict__ C,
                               int M, int N, int K) {
    int lane = threadIdx.x;
    int half = lane >> 4;      // 0 or 1
    int l16  = lane & 15;
    int n0 = (blockIdx.x * blockDim.y + threadIdx.y) * 32;
    int m0 = blockIdx.y * 32;

    const bf16_t* arow0 = A  + (size_t)(m0 + l16)      * K;
    const bf16_t* arow1 = A  + (size_t)(m0 + 16 + l16) * K;
    const bf16_t* brow0 = Bt + (size_t)(n0 + l16)      * K;
    const bf16_t* brow1 = Bt + (size_t)(n0 + 16 + l16) * K;

    v8f acc00 = {}, acc01 = {}, acc10 = {}, acc11 = {};
    for (int k0 = 0; k0 < K; k0 += 32) {
        v16bf a0 = load_frag(arow0, k0, half);
        v16bf a1 = load_frag(arow1, k0, half);
        v16bf b0 = load_frag(brow0, k0, half);
        v16bf b1 = load_frag(brow1, k0, half);
        acc00 = __builtin_amdgcn_wmma_f32_16x16x32_bf16(
                    false, a0, false, b0, (short)0, acc00, false, false);
        acc01 = __builtin_amdgcn_wmma_f32_16x16x32_bf16(
                    false, a0, false, b1, (short)0, acc01, false, false);
        acc10 = __builtin_amdgcn_wmma_f32_16x16x32_bf16(
                    false, a1, false, b0, (short)0, acc10, false, false);
        acc11 = __builtin_amdgcn_wmma_f32_16x16x32_bf16(
                    false, a1, false, b1, (short)0, acc11, false, false);
    }

    float bn0 = bias[n0 + l16];
    float bn1 = bias[n0 + 16 + l16];
    int   r0  = m0 + half * 8;         // rows for acc0x
    int   r1  = m0 + 16 + half * 8;    // rows for acc1x
#pragma unroll
    for (int r = 0; r < 8; ++r) {
        C[(size_t)(r0 + r) * N + n0 + l16]      = acc00[r] + bn0;
        C[(size_t)(r0 + r) * N + n0 + 16 + l16] = acc01[r] + bn1;
        C[(size_t)(r1 + r) * N + n0 + l16]      = acc10[r] + bn0;
        C[(size_t)(r1 + r) * N + n0 + 16 + l16] = acc11[r] + bn1;
    }
}

// ---------------------------------------------------------------------------
// Rank-1 outer attention: f[b,i] = resid[b,i] +
//     sum_j exp(s*q_i*k_j - m_i) * v_j / sum_j exp(s*q_i*k_j - m_i)
// m_i = (s*q_i >= 0) ? s*q_i*max(k) : s*q_i*min(k)  (exact row max of logits)
// One block (256 thr) per sample; k,v staged in LDS, read back as float4
// (ds_load_b128). exp-throughput bound (~150M v_exp_f32 total).
// ---------------------------------------------------------------------------
__global__ void attn_kernel(const float* __restrict__ q,
                            const float* __restrict__ k,
                            const float* __restrict__ v,
                            const float* __restrict__ resid,
                            bf16_t* __restrict__ fused,
                            int col_off) {
    __shared__ float sk[E_DIM];
    __shared__ float sv[E_DIM];
    __shared__ float rmax[256];
    __shared__ float rmin[256];

    const float scale = 0.05103103630798288f;   // 384^-0.5
    int b   = blockIdx.x;
    int tid = threadIdx.x;

    for (int j = tid; j < E_DIM; j += 256) {
        sk[j] = k[(size_t)b * E_DIM + j];
        sv[j] = v[(size_t)b * E_DIM + j];
    }
    __syncthreads();

    float mx = -3.4e38f, mn = 3.4e38f;
    for (int j = tid; j < E_DIM; j += 256) {
        mx = fmaxf(mx, sk[j]);
        mn = fminf(mn, sk[j]);
    }
    rmax[tid] = mx; rmin[tid] = mn;
    __syncthreads();
    for (int s = 128; s > 0; s >>= 1) {
        if (tid < s) {
            rmax[tid] = fmaxf(rmax[tid], rmax[tid + s]);
            rmin[tid] = fminf(rmin[tid], rmin[tid + s]);
        }
        __syncthreads();
    }
    float kmax = rmax[0], kmin = rmin[0];

    const float4* sk4 = (const float4*)sk;
    const float4* sv4 = (const float4*)sv;

    for (int i = tid; i < E_DIM; i += 256) {
        float qi = q[(size_t)b * E_DIM + i] * scale;
        float m  = (qi >= 0.f) ? qi * kmax : qi * kmin;
        float num = 0.f, den = 0.f;
#pragma unroll 2
        for (int j4 = 0; j4 < E_DIM / 4; ++j4) {
            float4 kk = sk4[j4];
            float4 vv = sv4[j4];
            float e0 = __expf(qi * kk.x - m);
            float e1 = __expf(qi * kk.y - m);
            float e2 = __expf(qi * kk.z - m);
            float e3 = __expf(qi * kk.w - m);
            num += e0 * vv.x + e1 * vv.y + e2 * vv.z + e3 * vv.w;
            den += e0 + e1 + e2 + e3;
        }
        float f = num / den + resid[(size_t)b * E_DIM + i];
        fused[(size_t)b * TWO_E + col_off + i] = (bf16_t)f;
    }
}

// ---------------------------------------------------------------------------
// Host-side orchestration
// ---------------------------------------------------------------------------
extern "C" void kernel_launch(void* const* d_in, const int* in_sizes, int n_in,
                              void* d_out, int out_size, void* d_ws, size_t ws_size,
                              hipStream_t stream) {
    (void)in_sizes; (void)n_in; (void)out_size; (void)ws_size;

    const float* x       = (const float*)d_in[0];
    const float* x_bpf   = (const float*)d_in[1];
    const float* gamma   = (const float*)d_in[2];
    const float* beta    = (const float*)d_in[3];
    const float* gamma_b = (const float*)d_in[4];
    const float* beta_b  = (const float*)d_in[5];
    const float* Wq      = (const float*)d_in[6];
    const float* bq      = (const float*)d_in[7];
    const float* Wk      = (const float*)d_in[8];
    const float* bk      = (const float*)d_in[9];
    const float* Wv      = (const float*)d_in[10];
    const float* bv      = (const float*)d_in[11];
    const float* Wq_bpf  = (const float*)d_in[12];
    const float* bq_bpf  = (const float*)d_in[13];
    const float* Wk_bpf  = (const float*)d_in[14];
    const float* bk_bpf  = (const float*)d_in[15];
    const float* Wv_bpf  = (const float*)d_in[16];
    const float* bv_bpf  = (const float*)d_in[17];
    const float* Wo      = (const float*)d_in[18];
    const float* bo      = (const float*)d_in[19];
    float* out = (float*)d_out;

    // ---- workspace layout (all offsets 256B-aligned) ----
    const size_t HELEMS  = (size_t)B_DIM * E_DIM;         // 196608
    const size_t WELEMS  = (size_t)E_DIM * E_DIM;         // 147456
    const size_t WOELEMS = (size_t)TWO_E * H_DIM;         // 393216
    bf16_t* h    = (bf16_t*)d_ws;
    bf16_t* hb   = h  + HELEMS;
    bf16_t* Wt   = hb + HELEMS;          // 6 x [E][E] (transposed, bf16)
    bf16_t* Wot  = Wt + 6 * WELEMS;      // [H][2E]  (transposed, bf16)
    float*  qbuf = (float*)(Wot + WOELEMS);
    float*  kbuf = qbuf + HELEMS;
    float*  vbuf = kbuf + HELEMS;
    float*  qbbf = vbuf + HELEMS;
    float*  kbbf = qbbf + HELEMS;
    float*  vbbf = kbbf + HELEMS;
    bf16_t* fused = (bf16_t*)(vbbf + HELEMS);  // [B][2E]

    // ---- 1) weight conversion / transpose ----
    {
        int cnt = E_DIM * E_DIM;
        dim3 blk(256), grd((cnt + 255) / 256);
        convert_transpose_kernel<<<grd, blk, 0, stream>>>(Wq,     Wt + 0 * WELEMS, E_DIM, E_DIM);
        convert_transpose_kernel<<<grd, blk, 0, stream>>>(Wk,     Wt + 1 * WELEMS, E_DIM, E_DIM);
        convert_transpose_kernel<<<grd, blk, 0, stream>>>(Wv,     Wt + 2 * WELEMS, E_DIM, E_DIM);
        convert_transpose_kernel<<<grd, blk, 0, stream>>>(Wq_bpf, Wt + 3 * WELEMS, E_DIM, E_DIM);
        convert_transpose_kernel<<<grd, blk, 0, stream>>>(Wk_bpf, Wt + 4 * WELEMS, E_DIM, E_DIM);
        convert_transpose_kernel<<<grd, blk, 0, stream>>>(Wv_bpf, Wt + 5 * WELEMS, E_DIM, E_DIM);
        int cnto = TWO_E * H_DIM;
        convert_transpose_kernel<<<dim3((cnto + 255) / 256), blk, 0, stream>>>(Wo, Wot, TWO_E, H_DIM);
    }

    // ---- 2) groupnorm (both branches) ----
    groupnorm_kernel<<<dim3(B_DIM / 4, 2), dim3(32, 4), 0, stream>>>(
        x, x_bpf, gamma, beta, gamma_b, beta_b, h, hb);

    // ---- 3) QKV GEMMs (bf16 WMMA, f32 accumulate), 32x32 tile per wave ----
    {
        dim3 blk(32, 4);
        dim3 grd(E_DIM / 128, B_DIM / 32);   // (3, 16)
        wmma_gemm_bf16<<<grd, blk, 0, stream>>>(h,  Wt + 0 * WELEMS, bq,     qbuf, B_DIM, E_DIM, E_DIM);
        wmma_gemm_bf16<<<grd, blk, 0, stream>>>(h,  Wt + 1 * WELEMS, bk,     kbuf, B_DIM, E_DIM, E_DIM);
        wmma_gemm_bf16<<<grd, blk, 0, stream>>>(h,  Wt + 2 * WELEMS, bv,     vbuf, B_DIM, E_DIM, E_DIM);
        wmma_gemm_bf16<<<grd, blk, 0, stream>>>(hb, Wt + 3 * WELEMS, bq_bpf, qbbf, B_DIM, E_DIM, E_DIM);
        wmma_gemm_bf16<<<grd, blk, 0, stream>>>(hb, Wt + 4 * WELEMS, bk_bpf, kbbf, B_DIM, E_DIM, E_DIM);
        wmma_gemm_bf16<<<grd, blk, 0, stream>>>(hb, Wt + 5 * WELEMS, bv_bpf, vbbf, B_DIM, E_DIM, E_DIM);
    }

    // ---- 4) rank-1 outer attention (exp-bound), writes fused [B, 2E] ----
    // branch 1: q_bpf attends over (k, v) from x; residual x -> cols [0, E)
    attn_kernel<<<dim3(B_DIM), dim3(256), 0, stream>>>(qbbf, kbuf, vbuf, x,     fused, 0);
    // branch 2: q attends over (k_bpf, v_bpf); residual x_bpf -> cols [E, 2E)
    attn_kernel<<<dim3(B_DIM), dim3(256), 0, stream>>>(qbuf, kbbf, vbbf, x_bpf, fused, E_DIM);

    // ---- 5) output GEMM: [B,2E] x [2E,H] + bo -> d_out ----
    {
        dim3 blk(32, 4);
        dim3 grd(H_DIM / 128, B_DIM / 32);   // (4, 16)
        wmma_gemm_bf16<<<grd, blk, 0, stream>>>(fused, Wot, bo, out, B_DIM, H_DIM, TWO_E);
    }
}
